// ConvBinLayer_15066745274548
// MI455X (gfx1250) — matrix operand order repaired and checked by reading, not compiled, not run on hardware
//
#include <hip/hip_runtime.h>
#include <stdint.h>

typedef __bf16 bf16_t;
typedef __attribute__((ext_vector_type(16))) __bf16 v16bf;
typedef __attribute__((ext_vector_type(8)))  float  v8f;

#define NBATCH  32
#define C_IN    256
#define K_OUT   256
#define H_IN    56
#define W_IN    56
#define H_OUT   54
#define W_OUT   54
#define RED     (C_IN * 9)        // 2304 reduction length
#define KCHUNKS (RED / 32)        // 72 chunks of K=32
#define GRP     8                 // K-chunks per LDS fragment group
#define NGRP    (KCHUNKS / GRP)   // 9 groups

// ---------------------------------------------------------------------------
// Prep: binarize weights (OIHW f32) into bf16 WMMA A-fragments.
// Layout: [mTile(16)][kChunk(72)][lane(32)][slot(16)] bf16 -> 32B/lane/chunk.
// A-matrix 16-bit layout (ISA 7.12.2): lane = half*16 + M%16,
//   slot<8  -> K = half*8 + slot ; slot>=8 -> K = 8 + half*8 + slot
// ---------------------------------------------------------------------------
__global__ void prep_wfrag(const float* __restrict__ w, bf16_t* __restrict__ frag) {
    int tid  = blockIdx.x * blockDim.x + threadIdx.x;     // 0 .. 589823
    int slot = tid & 15;
    int lane = (tid >> 4) & 31;
    int kc   = (tid >> 9) % KCHUNKS;
    int mt   = tid / (512 * KCHUNKS);
    int half = lane >> 4;
    int m    = mt * 16 + (lane & 15);
    int kin  = (slot < 8) ? (half * 8 + slot) : (8 + half * 8 + slot);
    int r    = kc * 32 + kin;
    int c    = r / 9;
    int s    = r - c * 9;
    float wv = w[(size_t)(m * C_IN + c) * 9 + s];
    frag[tid] = (wv >= 0.0f) ? (bf16_t)1.0f : (bf16_t)(-1.0f);
}

// ---------------------------------------------------------------------------
// Main: implicit-GEMM binary conv, bf16 WMMA with hi/lo f32 splitting.
// Block = 512 threads = 16 waves; one block per (n, h, 16-wide w tile).
// Async-stage x slab + offset LUT; per group of 8 K-chunks the block builds
// shared hi/lo B-fragments in LDS once, then each wave (one 16-channel M
// tile) runs 16 WMMAs off them. Fully unrolled consume lets the scheduler
// pipeline ds_load_b128 fragment reads ahead of the WMMA chain.
// ---------------------------------------------------------------------------
__global__ __launch_bounds__(512) void binconv_wmma(
    const float* __restrict__ x,
    const v16bf* __restrict__ wfrag,
    float* __restrict__ out)
{
    __shared__ float    xs[C_IN * 3 * 18];                 // 55296 B
    __shared__ uint16_t lut[RED];                          // 4608 B
    __shared__ __align__(32) uint16_t fragHi[GRP * 512];   // 8192 B
    __shared__ __align__(32) uint16_t fragLo[GRP * 512];   // 8192 B

    int t  = blockIdx.x;
    int wt = t & 3;                 // w tile: w0 = 0,16,32,48
    int h  = (t >> 2) % H_OUT;
    int n  = t / (4 * H_OUT);
    int w0 = wt * 16;

    // ---- async global->LDS stage of x[n][:, h..h+2, w0..w0+17] ----
    // 13824 dwords / 512 threads = 27 iterations exactly (no divergence).
    const float* xbase = x + (size_t)n * C_IN * H_IN * W_IN;
    for (int idx = threadIdx.x; idx < C_IN * 54; idx += 512) {
        int c    = idx / 54;
        int rem  = idx - c * 54;
        int dy   = rem / 18;
        int dx   = rem - dy * 18;
        int wsrc = w0 + dx;
        if (wsrc > W_IN - 1) wsrc = W_IN - 1;   // clamped values feed only masked columns
        const float* g = &xbase[((size_t)c * H_IN + (h + dy)) * W_IN + wsrc];
        uint32_t ldsoff = (uint32_t)(size_t)&xs[idx];
        asm volatile("global_load_async_to_lds_b32 %0, %1, off"
                     :: "v"(ldsoff), "v"(g) : "memory");
    }
    // ---- im2col offset LUT (block-invariant): lut[r] = c*54 + dy*18 + dx ----
    for (int r = threadIdx.x; r < RED; r += 512) {
        int c = r / 9, s = r - c * 9;
        int dy = s / 3, dx = s - dy * 3;
        lut[r] = (uint16_t)(c * 54 + dy * 18 + dx);
    }
    asm volatile("s_wait_asynccnt 0x0" ::: "memory");
    __syncthreads();

    int wave = threadIdx.x >> 5;    // 0..15 == M tile (16 output channels)
    int lane = threadIdx.x & 31;
    int half = lane >> 4;
    int col  = lane & 15;

    // B-build role: 512 threads cover one chunk's 32 lanes x 16 slots
    int flane = threadIdx.x >> 4;   // fragment lane 0..31
    int fslot = threadIdx.x & 15;   // fragment slot 0..15
    int fhalf = flane >> 4;
    int fcol  = flane & 15;

    v8f acc = {};
    const v16bf* afrag = wfrag + (size_t)wave * KCHUNKS * 32 + lane;

    for (int g = 0; g < NGRP; ++g) {
        // ---- cooperative B-fragment build for GRP chunks (done ONCE) ----
        // B layout: slot<8 -> K = half*8 + slot ; slot>=8 -> K = 16 + half*8 + (slot-8).
        int rb = (g * GRP) * 32 + fhalf * 8 + fslot + (fslot & 8);
#pragma unroll
        for (int j = 0; j < GRP; ++j) {
            int off = (int)lut[rb + j * 32] + fcol;
            float v = xs[off];
            uint32_t bits = __builtin_bit_cast(uint32_t, v);
            uint32_t hibits = bits & 0xffff0000u;         // truncating bf16 split
            float hv = __builtin_bit_cast(float, hibits);
            fragHi[j * 512 + threadIdx.x] = (uint16_t)(bits >> 16);
            bf16_t lo = (bf16_t)(v - hv);                 // residual bf16 part
            fragLo[j * 512 + threadIdx.x] = __builtin_bit_cast(uint16_t, lo);
        }
        __syncthreads();

        // ---- consume: 16 WMMAs per wave per group off shared fragments ----
#pragma unroll
        for (int j = 0; j < GRP; ++j) {
            int kc = g * GRP + j;
            v16bf a   = afrag[(size_t)kc * 32];           // 32B, L2-resident
            v16bf bhi = *(const v16bf*)&fragHi[j * 512 + lane * 16];
            v16bf blo = *(const v16bf*)&fragLo[j * 512 + lane * 16];
            acc = __builtin_amdgcn_wmma_f32_16x16x32_bf16(
                      false, a, false, bhi, (short)0, acc, false, false);
            acc = __builtin_amdgcn_wmma_f32_16x16x32_bf16(
                      false, a, false, blo, (short)0, acc, false, false);
        }
        __syncthreads();
    }

    // D layout: lanes 0-15 -> M=j, lanes 16-31 -> M=8+j ; N = col
    int wcol = w0 + col;
    if (wcol < W_OUT) {
        int ch0 = wave * 16 + half * 8;
        size_t base = (((size_t)n * K_OUT + ch0) * H_OUT + h) * W_OUT + wcol;
#pragma unroll
        for (int j = 0; j < 8; ++j)
            out[base + (size_t)j * H_OUT * W_OUT] = acc[j];
    }
}

// ---------------------------------------------------------------------------
extern "C" void kernel_launch(void* const* d_in, const int* in_sizes, int n_in,
                              void* d_out, int out_size, void* d_ws, size_t ws_size,
                              hipStream_t stream) {
    const float* x = (const float*)d_in[0];   // (32,256,56,56)
    const float* w = (const float*)d_in[1];   // (256,256,3,3)
    float* out = (float*)d_out;               // (32,256,54,54)

    bf16_t* frag = (bf16_t*)d_ws;             // 589824 * 2 B = 1.18 MB

    prep_wfrag<<<(16 * KCHUNKS * 512) / 256, 256, 0, stream>>>(w, frag);

    int blocks = NBATCH * H_OUT * 4;          // 6912
    binconv_wmma<<<blocks, 512, 0, stream>>>(x, (const v16bf*)frag, out);
}
